// EdgePredictionGNN_82162724372639
// MI455X (gfx1250) — matrix operand order repaired and checked by reading, not compiled, not run on hardware
//
#include <hip/hip_runtime.h>
#include <hip/hip_bf16.h>
#include <stdint.h>

typedef __attribute__((ext_vector_type(16))) _Float16 v16h;
typedef __attribute__((ext_vector_type(8)))  float    v8f;

#define NEG_SLOPE 0.2f

// ---------------- order-preserving float<->int key for atomicMax -----------
__device__ __forceinline__ int f2key(float f) {
  int b = __float_as_int(f);
  return b >= 0 ? b : (b ^ 0x7fffffff);
}
__device__ __forceinline__ float key2f(int k) {
  int b = k >= 0 ? k : (k ^ 0x7fffffff);
  return __int_as_float(b);
}

// ---------------- WMMA fragment loaders ------------------------------------
// A fragment 16x32 f16 (M = lane&15). Per CDNA5 ISA layout:
// lanes 0-15: j=0..7 -> K=j, j=8..15 -> K=16+(j-8); lanes 16-31: +8.
// Both runs are 8 contiguous floats -> vectorizable global loads.
__device__ __forceinline__ v16h load_a_frag(const float* __restrict__ rowptr, int k0, int lane) {
  const float* pp = rowptr + k0 + ((lane & 16) ? 8 : 0);
  v16h a;
#pragma unroll
  for (int j = 0; j < 8; ++j) {
    a[j]     = (_Float16)pp[j];
    a[8 + j] = (_Float16)pp[16 + j];
  }
  return a;
}

// B fragment: prepacked wave-striped layout, one contiguous 32B load per lane.
__device__ __forceinline__ v16h load_b_frag(const _Float16* __restrict__ pack, int kt, int nt, int lane) {
  return ((const v16h*)pack)[(kt * 4 + nt) * 32 + lane];
}

// Pack a 64x64 (row-major, ld=64) f32 weight block starting at row `row_off`
// into the B-fragment layout: element j of lane holds K = kt*32 + (lane<16?0:8) + (j&7) + (j<8?0:16),
// N = nt*16 + (lane&15). 4096 halves total.
__global__ void pack_b_kernel(const float* __restrict__ W, int row_off, _Float16* __restrict__ out) {
  int idx = blockIdx.x * blockDim.x + threadIdx.x;
  if (idx >= 4096) return;
  int j    = idx & 15;
  int lane = (idx >> 4) & 31;
  int nt   = (idx >> 9) & 3;
  int kt   = (idx >> 11) & 1;
  int K = kt * 32 + ((lane & 16) ? 8 : 0) + (j & 7) + ((j & 8) ? 16 : 0);
  int Nn = nt * 16 + (lane & 15);
  out[idx] = (_Float16)W[(size_t)(row_off + K) * 64 + Nn];
}

// ce[L] = dot(We_L, ae_L)  (ED==1 so alpha_edge = ea * ce)
__global__ void ce_kernel(const float* We1, const float* ae1,
                          const float* We2, const float* ae2, float* ce) {
  float s1 = 0.f, s2 = 0.f;
  for (int k = 0; k < 64; ++k) { s1 += We1[k] * ae1[k]; s2 += We2[k] * ae2[k]; }
  ce[0] = s1; ce[1] = s2;
}

// ---------------- self-loop attr = segment mean over dst --------------------
__global__ void loop_sum_kernel(const int* __restrict__ dst, const float* __restrict__ ea,
                                float* __restrict__ lsum, float* __restrict__ cnt, int E_) {
  int e = blockIdx.x * blockDim.x + threadIdx.x;
  if (e >= E_) return;
  int d = dst[e];
  atomicAdd(&lsum[d], ea[e]);
  atomicAdd(&cnt[d], 1.0f);
}
__global__ void loop_div_kernel(float* __restrict__ lsum, const float* __restrict__ cnt, int n) {
  int i = blockIdx.x * blockDim.x + threadIdx.x;
  if (i >= n) return;
  lsum[i] = lsum[i] / fmaxf(cnt[i], 1.0f);
}

// ---------------- h = X @ W via WMMA (one wave per 16-row tile) -------------
__global__ void gemm64_kernel(const float* __restrict__ X, const _Float16* __restrict__ Bp,
                              float* __restrict__ H, int nrows) {
  int lane = threadIdx.x & 31;
  int tile = blockIdx.x * 8 + (threadIdx.x >> 5);
  if (tile * 16 >= nrows) return;
  int row0 = tile * 16;
  const float* rp = X + (size_t)(row0 + (lane & 15)) * 64;
  v16h a0 = load_a_frag(rp, 0, lane);
  v16h a1 = load_a_frag(rp, 32, lane);
  int mofs = (lane & 16) ? 8 : 0;
#pragma unroll
  for (int nt = 0; nt < 4; ++nt) {
    v8f c = {};
    c = __builtin_amdgcn_wmma_f32_16x16x32_f16(false, a0, false, load_b_frag(Bp, 0, nt, lane),
                                               (short)0, c, false, false);
    c = __builtin_amdgcn_wmma_f32_16x16x32_f16(false, a1, false, load_b_frag(Bp, 1, nt, lane),
                                               (short)0, c, false, false);
    float* hp = H + (size_t)(row0 + mofs) * 64 + nt * 16 + (lane & 15);
#pragma unroll
    for (int r = 0; r < 8; ++r) hp[(size_t)r * 64] = c[r];
  }
}

// ---------------- per-node attention dots: as[n]=h.a_s, ad[n]=h.a_d ---------
__global__ void alpha_node_kernel(const float* __restrict__ H, const float* __restrict__ a_s,
                                  const float* __restrict__ a_d, float* __restrict__ outs,
                                  float* __restrict__ outd, int n) {
  int lane = threadIdx.x & 31;
  int node = blockIdx.x * 8 + (threadIdx.x >> 5);
  if (node >= n) return;
  float h0 = H[(size_t)node * 64 + lane];
  float h1 = H[(size_t)node * 64 + 32 + lane];
  float s = h0 * a_s[lane] + h1 * a_s[lane + 32];
  float d = h0 * a_d[lane] + h1 * a_d[lane + 32];
#pragma unroll
  for (int m = 16; m >= 1; m >>= 1) {
    s += __shfl_xor(s, m);
    d += __shfl_xor(d, m);
  }
  if (lane == 0) { outs[node] = s; outd[node] = d; }
}

__global__ void init_mkey_kernel(int* __restrict__ mkey, int n) {
  int i = blockIdx.x * blockDim.x + threadIdx.x;
  if (i < n) mkey[i] = (int)0x80000000;
}

// alpha = leaky_relu(as[src] + ad[dst] + ea*ce); segment max via int-key atomicMax
__global__ void edge_alpha_kernel(const int* __restrict__ src, const int* __restrict__ dst,
                                  const float* __restrict__ ea, const float* __restrict__ loop_attr,
                                  const float* __restrict__ asn, const float* __restrict__ adn,
                                  const float* __restrict__ cep, float* __restrict__ alpha,
                                  int* __restrict__ mkey, int E_, int total) {
  int e = blockIdx.x * blockDim.x + threadIdx.x;
  if (e >= total) return;
  int s, d; float a;
  if (e < E_) { s = src[e]; d = dst[e]; a = ea[e]; }
  else        { s = d = e - E_; a = loop_attr[s]; }
  float al = asn[s] + adn[d] + a * (*cep);
  al = al > 0.f ? al : NEG_SLOPE * al;
  alpha[e] = al;
  atomicMax(&mkey[d], f2key(al));
}

// ex = exp(alpha - m[dst]); den[dst] += ex
__global__ void edge_exp_kernel(const int* __restrict__ dst, const int* __restrict__ mkey,
                                float* __restrict__ alpha, float* __restrict__ den,
                                int E_, int total) {
  int e = blockIdx.x * blockDim.x + threadIdx.x;
  if (e >= total) return;
  int d = (e < E_) ? dst[e] : (e - E_);
  float ex = __expf(alpha[e] - key2f(mkey[d]));
  alpha[e] = ex;
  atomicAdd(&den[d], ex);
}

// accum[dst] += (ex/den[dst]) * h[src]  (one wave per edge, 2 cols per lane)
__global__ void edge_aggr_kernel(const float* __restrict__ ex, const float* __restrict__ den,
                                 const float* __restrict__ H, const int* __restrict__ src,
                                 const int* __restrict__ dst, float* __restrict__ accum,
                                 int E_, int total) {
  int lane = threadIdx.x & 31;
  int e = blockIdx.x * 8 + (threadIdx.x >> 5);
  if (e >= total) return;
  int s, d;
  if (e < E_) { s = src[e]; d = dst[e]; } else { s = d = e - E_; }
  float w = ex[e] / den[d];
  float h0 = H[(size_t)s * 64 + lane];
  float h1 = H[(size_t)s * 64 + 32 + lane];
  atomicAdd(&accum[(size_t)d * 64 + lane], w * h0);
  atomicAdd(&accum[(size_t)d * 64 + 32 + lane], w * h1);
}

// h_out = relu(accum + b), in place
__global__ void finalize_kernel(float* __restrict__ accum, const float* __restrict__ b, int n) {
  int i = blockIdx.x * blockDim.x + threadIdx.x;
  if (i >= n * 64) return;
  float v = accum[i] + b[i & 63];
  accum[i] = v > 0.f ? v : 0.f;
}

// ---------------- edge MLP: relu([h_s, ea, h_d]@Wm1 + bm1) @ Wm2 + bm2 ------
// One wave per 16-edge tile; gathered-row WMMA for the two 64-col blocks.
__global__ void edge_mlp_kernel(const float* __restrict__ H, const int* __restrict__ src,
                                const int* __restrict__ dst, const float* __restrict__ ea,
                                const _Float16* __restrict__ Bs, const _Float16* __restrict__ Bd,
                                const float* __restrict__ Wm1row, const float* __restrict__ bm1,
                                const float* __restrict__ Wm2, const float* __restrict__ bm2,
                                float* __restrict__ out, int nedges) {
  int lane = threadIdx.x & 31;
  int tile = blockIdx.x * 8 + (threadIdx.x >> 5);
  if (tile * 16 >= nedges) return;
  int e0 = tile * 16;
  int m = lane & 15;
  int rs = src[e0 + m];
  int rd = dst[e0 + m];
  v16h as0 = load_a_frag(H + (size_t)rs * 64, 0, lane);
  v16h as1 = load_a_frag(H + (size_t)rs * 64, 32, lane);
  v16h ad0 = load_a_frag(H + (size_t)rd * 64, 0, lane);
  v16h ad1 = load_a_frag(H + (size_t)rd * 64, 32, lane);
  int mofs = (lane & 16) ? 8 : 0;
  float eav[8], rowacc[8];
#pragma unroll
  for (int r = 0; r < 8; ++r) {
    eav[r] = ea[e0 + mofs + r];
    rowacc[r] = 0.f;
  }
#pragma unroll
  for (int nt = 0; nt < 4; ++nt) {
    v8f c = {};
    c = __builtin_amdgcn_wmma_f32_16x16x32_f16(false, as0, false, load_b_frag(Bs, 0, nt, lane),
                                               (short)0, c, false, false);
    c = __builtin_amdgcn_wmma_f32_16x16x32_f16(false, as1, false, load_b_frag(Bs, 1, nt, lane),
                                               (short)0, c, false, false);
    c = __builtin_amdgcn_wmma_f32_16x16x32_f16(false, ad0, false, load_b_frag(Bd, 0, nt, lane),
                                               (short)0, c, false, false);
    c = __builtin_amdgcn_wmma_f32_16x16x32_f16(false, ad1, false, load_b_frag(Bd, 1, nt, lane),
                                               (short)0, c, false, false);
    int col = nt * 16 + m;
    float wr = Wm1row[col];
    float bc = bm1[col];
    float w2 = Wm2[col];
#pragma unroll
    for (int r = 0; r < 8; ++r) {
      float z = c[r] + eav[r] * wr + bc;
      z = z > 0.f ? z : 0.f;
      rowacc[r] += z * w2;
    }
  }
  float b2v = *bm2;
#pragma unroll
  for (int r = 0; r < 8; ++r) {
    float v = rowacc[r];
    v += __shfl_xor(v, 1);
    v += __shfl_xor(v, 2);
    v += __shfl_xor(v, 4);
    v += __shfl_xor(v, 8);
    if (m == 0) out[e0 + mofs + r] = v + b2v;
  }
}

extern "C" void kernel_launch(void* const* d_in, const int* in_sizes, int n_in,
                              void* d_out, int out_size, void* d_ws, size_t ws_size,
                              hipStream_t stream) {
  const float* x   = (const float*)d_in[0];
  const int*   ei  = (const int*)d_in[1];
  const float* ea  = (const float*)d_in[2];
  const float* W1  = (const float*)d_in[3];
  const float* as1 = (const float*)d_in[4];
  const float* ad1 = (const float*)d_in[5];
  const float* We1 = (const float*)d_in[6];
  const float* ae1 = (const float*)d_in[7];
  const float* b1  = (const float*)d_in[8];
  const float* W2  = (const float*)d_in[9];
  const float* as2 = (const float*)d_in[10];
  const float* ad2 = (const float*)d_in[11];
  const float* We2 = (const float*)d_in[12];
  const float* ae2 = (const float*)d_in[13];
  const float* b2  = (const float*)d_in[14];
  const float* Wm1 = (const float*)d_in[15];
  const float* bm1 = (const float*)d_in[16];
  const float* Wm2 = (const float*)d_in[17];
  const float* bm2 = (const float*)d_in[18];
  (void)n_in; (void)out_size; (void)ws_size;

  const int N = in_sizes[0] / 64;   // 100000 (multiple of 16)
  const int E = in_sizes[1] / 2;    // 1600000 (multiple of 16)
  const int T = E + N;
  const int* srcp = ei;
  const int* dstp = ei + E;

  uintptr_t p = (uintptr_t)d_ws;
  auto carve = [&](size_t bytes) -> void* {
    p = (p + 255) & ~(uintptr_t)255;
    void* r = (void*)p;
    p += bytes;
    return r;
  };
  float* hP   = (float*)carve((size_t)N * 64 * 4);  // transformed features
  float* hQ   = (float*)carve((size_t)N * 64 * 4);  // accum / layer output
  float* alph = (float*)carve((size_t)T * 4);
  float* asn  = (float*)carve((size_t)N * 4);
  float* adn  = (float*)carve((size_t)N * 4);
  float* den  = (float*)carve((size_t)N * 4);
  float* lsum = (float*)carve((size_t)N * 4);       // becomes loop_attr
  float* cnt  = (float*)carve((size_t)N * 4);
  int*   mkey = (int*)carve((size_t)N * 4);
  _Float16* W1p = (_Float16*)carve(4096 * 2);
  _Float16* W2p = (_Float16*)carve(4096 * 2);
  _Float16* Wap = (_Float16*)carve(4096 * 2);       // Wm1 rows 0..63
  _Float16* Wbp = (_Float16*)carve(4096 * 2);       // Wm1 rows 65..128
  float* ce = (float*)carve(2 * 4);

  // one-time packing / scalars
  pack_b_kernel<<<16, 256, 0, stream>>>(W1, 0, W1p);
  pack_b_kernel<<<16, 256, 0, stream>>>(W2, 0, W2p);
  pack_b_kernel<<<16, 256, 0, stream>>>(Wm1, 0, Wap);
  pack_b_kernel<<<16, 256, 0, stream>>>(Wm1, 65, Wbp);
  ce_kernel<<<1, 1, 0, stream>>>(We1, ae1, We2, ae2, ce);

  // self-loop edge_attr (mean of incoming per node)
  hipMemsetAsync(lsum, 0, (size_t)N * 4, stream);
  hipMemsetAsync(cnt, 0, (size_t)N * 4, stream);
  loop_sum_kernel<<<(E + 255) / 256, 256, 0, stream>>>(dstp, ea, lsum, cnt, E);
  loop_div_kernel<<<(N + 255) / 256, 256, 0, stream>>>(lsum, cnt, N);

  const float* xin = x;
  const _Float16* Wp[2] = {W1p, W2p};
  const float* avs[2] = {as1, as2};
  const float* avd[2] = {ad1, ad2};
  const float* bb[2]  = {b1, b2};
  for (int L = 0; L < 2; ++L) {
    gemm64_kernel<<<(N / 16 + 7) / 8, 256, 0, stream>>>(xin, Wp[L], hP, N);
    alpha_node_kernel<<<(N + 7) / 8, 256, 0, stream>>>(hP, avs[L], avd[L], asn, adn, N);
    init_mkey_kernel<<<(N + 255) / 256, 256, 0, stream>>>(mkey, N);
    hipMemsetAsync(den, 0, (size_t)N * 4, stream);
    hipMemsetAsync(hQ, 0, (size_t)N * 64 * 4, stream);  // ordered after gemm read of hQ
    edge_alpha_kernel<<<(T + 255) / 256, 256, 0, stream>>>(srcp, dstp, ea, lsum, asn, adn,
                                                           ce + L, alph, mkey, E, T);
    edge_exp_kernel<<<(T + 255) / 256, 256, 0, stream>>>(dstp, mkey, alph, den, E, T);
    edge_aggr_kernel<<<(T + 7) / 8, 256, 0, stream>>>(alph, den, hP, srcp, dstp, hQ, E, T);
    finalize_kernel<<<(N * 64 + 255) / 256, 256, 0, stream>>>(hQ, bb[L], N);
    xin = hQ;  // layer output feeds next layer
  }

  // edge MLP on h2 (= hQ)
  edge_mlp_kernel<<<(E / 16 + 7) / 8, 256, 0, stream>>>(hQ, srcp, dstp, ea, Wap, Wbp,
                                                        Wm1 + 64 * 64, bm1, Wm2, bm2,
                                                        (float*)d_out, E);
}
